// Net_7172595384447
// MI455X (gfx1250) — compile-verified, matrix operand order checked by reading, not run on hardware
//
#include <hip/hip_runtime.h>
#include <hip/hip_bf16.h>

typedef __attribute__((ext_vector_type(16))) _Float16 v16h;
typedef __attribute__((ext_vector_type(8)))  float    v8f;

// ---------------------------------------------------------------------------
// Pack W[K,F] (fp32, row-major) into per-lane WMMA B-fragment layout (f16).
// Fragment layout (CDNA5 ISA, 16-bit B 32x16):
//   lane l -> col l%16 ; element e -> K = (e>=8?16:0) + ((l>=16)?8:0) + (e&7)
// Packed index: (((kti*(F/16) + t) * 32 + l) * 16 + e)  -> 32 B per lane/tile.
// ---------------------------------------------------------------------------
__global__ void pack_w_kernel(const float* __restrict__ W,
                              _Float16* __restrict__ Wp, int K, int F) {
  const int NT = F / 16;
  const int total = (K / 32) * NT * 32;
  int i = blockIdx.x * blockDim.x + threadIdx.x;
  if (i >= total) return;
  const int l   = i & 31;
  const int t   = (i >> 5) % NT;
  const int kti = (i >> 5) / NT;
  const int col = l & 15;
  const int kg  = (l >> 4) << 3;
  _Float16* out = Wp + (size_t)i * 16;
#pragma unroll
  for (int e = 0; e < 16; ++e) {
    const int k = ((e >= 8) ? 16 : 0) + kg + (e & 7);
    out[e] = (_Float16)W[(size_t)(kti * 32 + k) * F + t * 16 + col];
  }
}

// ---------------------------------------------------------------------------
// WMMA GEMM:  H[N,F] = X[N,K] @ W[K,F]   (X fp32, W pre-packed f16, f32 acc)
// One wave computes a 16-row strip across all F columns (NT = F/16 tiles).
// A fetched as 4x global_load_b128; all NT B fragments preloaded into
// independent registers so loads clause up and overlap the WMMA chain.
// ---------------------------------------------------------------------------
template <int NT>
__global__ void __launch_bounds__(256)
gemm_wmma_kernel(const float* __restrict__ X,
                 const _Float16* __restrict__ Wp,
                 float* __restrict__ H,
                 int nRowTiles, int K) {
  constexpr int F = NT * 16;
  const int rowTile = blockIdx.x * blockDim.y + threadIdx.y;
  if (rowTile >= nRowTiles) return;          // wave-uniform: EXEC stays all-1s

  const int l   = threadIdx.x;               // lane 0..31
  const int col = l & 15;                    // A-row / B-col / D-col index
  const int kg  = (l >> 4) << 3;             // 0 or 8

  const float* xr = X + (size_t)(rowTile * 16 + col) * K + kg;

  v8f zero = {};
  v8f acc[NT];
#pragma unroll
  for (int t = 0; t < NT; ++t) acc[t] = zero;

  const int nkt = K / 32;
  for (int kti = 0; kti < nkt; ++kti) {
    // ---- B fragments: independent regs -> one load clause, partial waits ----
    const _Float16* wb = Wp + (size_t)(kti * NT) * 512 + (size_t)l * 16;
    v16h bfrag[NT];
#pragma unroll
    for (int t = 0; t < NT; ++t)
      bfrag[t] = *(const v16h*)(wb + (size_t)t * 512);   // 32B aligned

    // ---- A fragment: two contiguous 8-float runs per lane ----
    const float* xp = xr + kti * 32;
    const float4 p0 = ((const float4*)xp)[0];        // K = kg + 0..3
    const float4 p1 = ((const float4*)xp)[1];        // K = kg + 4..7
    const float4 p2 = ((const float4*)(xp + 16))[0]; // K = 16 + kg + 0..3
    const float4 p3 = ((const float4*)(xp + 16))[1]; // K = 16 + kg + 4..7
    v16h a;
    a[0]  = (_Float16)p0.x; a[1]  = (_Float16)p0.y;
    a[2]  = (_Float16)p0.z; a[3]  = (_Float16)p0.w;
    a[4]  = (_Float16)p1.x; a[5]  = (_Float16)p1.y;
    a[6]  = (_Float16)p1.z; a[7]  = (_Float16)p1.w;
    a[8]  = (_Float16)p2.x; a[9]  = (_Float16)p2.y;
    a[10] = (_Float16)p2.z; a[11] = (_Float16)p2.w;
    a[12] = (_Float16)p3.x; a[13] = (_Float16)p3.y;
    a[14] = (_Float16)p3.z; a[15] = (_Float16)p3.w;

#pragma unroll
    for (int t = 0; t < NT; ++t) {
      acc[t] = __builtin_amdgcn_wmma_f32_16x16x32_f16(
          /*neg_a=*/false, a, /*neg_b=*/false, bfrag[t],
          /*c_mod=*/(short)0, acc[t], /*reuse_a=*/false, /*reuse_b=*/false);
    }
  }

  const int rbase = rowTile * 16 + kg;       // kg doubles as C/D row-half offset
#pragma unroll
  for (int t = 0; t < NT; ++t)
#pragma unroll
    for (int r = 0; r < 8; ++r)
      H[(size_t)(rbase + r) * F + t * 16 + col] = acc[t][r];
}

// ---------------------------------------------------------------------------
// Graph-normalization helpers
// ---------------------------------------------------------------------------
__global__ void zero_f32_kernel(float* __restrict__ p, int n) {
  int i = blockIdx.x * blockDim.x + threadIdx.x;
  if (i < n) p[i] = 0.0f;
}

__global__ void degree_kernel(const int* __restrict__ dst,
                              float* __restrict__ deg, int E) {
  int i = blockIdx.x * blockDim.x + threadIdx.x;
  if (i < E) atomicAdd(&deg[dst[i]], 1.0f);
}

__global__ void dinv_kernel(const float* __restrict__ deg,
                            float* __restrict__ dinv,
                            float* __restrict__ dinv2, int n) {
  int i = blockIdx.x * blockDim.x + threadIdx.x;
  if (i < n) {
    float d = deg[i] + 1.0f;            // + implicit self loop
    dinv[i]  = rsqrtf(d);
    dinv2[i] = 1.0f / d;
  }
}

__global__ void coef_kernel(const int* __restrict__ src,
                            const int* __restrict__ dst,
                            const float* __restrict__ dinv,
                            float* __restrict__ coef, int E) {
  int i = blockIdx.x * blockDim.x + threadIdx.x;
  if (i < E) coef[i] = dinv[src[i]] * dinv[dst[i]];
}

// A[i,f] = H[i,f] * dinv2[i]   (self-loop contribution, init of aggregator)
__global__ void selfinit_kernel(const float* __restrict__ H,
                                const float* __restrict__ dinv2,
                                float* __restrict__ A, int total, int F) {
  int i = blockIdx.x * blockDim.x + threadIdx.x;
  if (i < total) A[i] = H[i] * dinv2[i / F];
}

// per (edge, 4-feature chunk): A[dst] += H[src] * coef
template <int F>
__global__ void scatter_kernel(const float* __restrict__ H,
                               const int* __restrict__ src,
                               const int* __restrict__ dst,
                               const float* __restrict__ coef,
                               float* __restrict__ A, int E) {
  constexpr int CH = F / 4;
  long long t = (long long)blockIdx.x * blockDim.x + threadIdx.x;
  if (t >= (long long)E * CH) return;
  int e = (int)(t / CH);
  int c = (int)(t % CH);
  int s = src[e], d = dst[e];
  float cf = coef[e];
  float4 v = ((const float4*)(H + (size_t)s * F))[c];
  float* ap = A + (size_t)d * F + c * 4;
  atomicAdd(ap + 0, v.x * cf);
  atomicAdd(ap + 1, v.y * cf);
  atomicAdd(ap + 2, v.z * cf);
  atomicAdd(ap + 3, v.w * cf);
}

__global__ void bias_relu_kernel(const float* __restrict__ A,
                                 const float* __restrict__ b,
                                 float* __restrict__ out, int total, int F) {
  int i = blockIdx.x * blockDim.x + threadIdx.x;
  if (i < total) out[i] = fmaxf(A[i] + b[i % F], 0.0f);
}

// out[i,:] = log_softmax(0.2*(A2+b2) + 0.1*(Ad2+bd2))  over 16 classes
__global__ void final_kernel(const float* __restrict__ A2,
                             const float* __restrict__ b2,
                             const float* __restrict__ Ad2,
                             const float* __restrict__ bd2,
                             float* __restrict__ out, int n) {
  int i = blockIdx.x * blockDim.x + threadIdx.x;
  if (i >= n) return;
  float v[16];
  float mx = -3.4e38f;
#pragma unroll
  for (int c = 0; c < 16; ++c) {
    float val = 0.2f * (A2[(size_t)i * 16 + c] + b2[c]) +
                0.1f * (Ad2[(size_t)i * 16 + c] + bd2[c]);
    v[c] = val;
    mx = fmaxf(mx, val);
  }
  float s = 0.0f;
#pragma unroll
  for (int c = 0; c < 16; ++c) s += expf(v[c] - mx);
  float ls = logf(s);
#pragma unroll
  for (int c = 0; c < 16; ++c) out[(size_t)i * 16 + c] = v[c] - mx - ls;
}

// ---------------------------------------------------------------------------
// Launch
// ---------------------------------------------------------------------------
static inline int cdivll(long long a, long long b) { return (int)((a + b - 1) / b); }

extern "C" void kernel_launch(void* const* d_in, const int* in_sizes, int n_in,
                              void* d_out, int out_size, void* d_ws, size_t ws_size,
                              hipStream_t stream) {
  (void)n_in; (void)out_size; (void)ws_size;

  const float* x   = (const float*)d_in[0];   // [N, 512]
  const float* dat = (const float*)d_in[1];   // [N, 64]
  const int*   ei  = (const int*)  d_in[2];   // [2, E]
  const float* W1  = (const float*)d_in[3];   // [512, 64]
  const float* b1  = (const float*)d_in[4];   // [64]
  const float* W2  = (const float*)d_in[5];   // [64, 16]
  const float* b2  = (const float*)d_in[6];   // [16]
  const float* Wd1 = (const float*)d_in[7];   // [64, 32]
  const float* bd1 = (const float*)d_in[8];   // [32]
  const float* Wd2 = (const float*)d_in[9];   // [32, 16]
  const float* bd2 = (const float*)d_in[10];  // [16]
  float* out = (float*)d_out;                 // [N, 16]

  const int N = in_sizes[0] / 512;            // 50000
  const int E = in_sizes[2] / 2;              // 800000
  const int* src = ei;
  const int* dst = ei + E;

  // ---- workspace layout (256B aligned slices) ----
  char* base = (char*)d_ws;
  size_t off = 0;
  auto alloc = [&](size_t bytes) -> void* {
    void* p = (void*)(base + off);
    off += (bytes + 255) & ~(size_t)255;
    return p;
  };
  float* deg   = (float*)alloc((size_t)N * 4);
  float* dinv  = (float*)alloc((size_t)N * 4);
  float* dinv2 = (float*)alloc((size_t)N * 4);
  float* coef  = (float*)alloc((size_t)E * 4);
  float* G1  = (float*)alloc((size_t)N * 64 * 4);   // x  @ W1
  float* A1  = (float*)alloc((size_t)N * 64 * 4);   // aggregated
  float* h1  = (float*)alloc((size_t)N * 64 * 4);   // relu(A1 + b1)
  float* Gd  = (float*)alloc((size_t)N * 32 * 4);   // dat @ Wd1
  float* Ad  = (float*)alloc((size_t)N * 32 * 4);
  float* hd1 = (float*)alloc((size_t)N * 32 * 4);
  float* G2  = (float*)alloc((size_t)N * 16 * 4);   // h1 @ W2
  float* A2  = (float*)alloc((size_t)N * 16 * 4);
  float* Gd2 = (float*)alloc((size_t)N * 16 * 4);   // hd1 @ Wd2
  float* Ad2 = (float*)alloc((size_t)N * 16 * 4);
  // packed f16 weight fragments: (K/32)*(F/16)*32 lanes * 16 halves
  _Float16* Wp1  = (_Float16*)alloc((size_t)(512/32) * (64/16) * 512 * 2);
  _Float16* Wpd1 = (_Float16*)alloc((size_t)(64/32)  * (32/16) * 512 * 2);
  _Float16* Wp2  = (_Float16*)alloc((size_t)(64/32)  * (16/16) * 512 * 2);
  _Float16* Wpd2 = (_Float16*)alloc((size_t)(32/32)  * (16/16) * 512 * 2);

  const int T = 256;
  const int nRowTiles = N / 16;             // N = 50000 = 16 * 3125
  dim3 gblk(32, 8);                         // 8 waves per block, 1 row-tile each
  const int ggrid = cdivll(nRowTiles, 8);

  // ---- pack weights into WMMA fragment layout (f16) ----
  pack_w_kernel<<<cdivll((512/32)*(64/16)*32, T), T, 0, stream>>>(W1,  Wp1,  512, 64);
  pack_w_kernel<<<cdivll((64/32)*(32/16)*32,  T), T, 0, stream>>>(Wd1, Wpd1, 64,  32);
  pack_w_kernel<<<cdivll((64/32)*(16/16)*32,  T), T, 0, stream>>>(W2,  Wp2,  64,  16);
  pack_w_kernel<<<cdivll((32/32)*(16/16)*32,  T), T, 0, stream>>>(Wd2, Wpd2, 32,  16);

  // ---- graph normalization (shared by all 4 convs) ----
  zero_f32_kernel<<<cdivll(N, T), T, 0, stream>>>(deg, N);
  degree_kernel  <<<cdivll(E, T), T, 0, stream>>>(dst, deg, E);
  dinv_kernel    <<<cdivll(N, T), T, 0, stream>>>(deg, dinv, dinv2, N);
  coef_kernel    <<<cdivll(E, T), T, 0, stream>>>(src, dst, dinv, coef, E);

  // ---- conv1: h1 = relu(agg(x @ W1) + b1), F=64, K=512 ----
  gemm_wmma_kernel<4><<<ggrid, gblk, 0, stream>>>(x, Wp1, G1, nRowTiles, 512);
  selfinit_kernel<<<cdivll((long long)N * 64, T), T, 0, stream>>>(G1, dinv2, A1, N * 64, 64);
  scatter_kernel<64><<<cdivll((long long)E * 16, T), T, 0, stream>>>(G1, src, dst, coef, A1, E);
  bias_relu_kernel<<<cdivll((long long)N * 64, T), T, 0, stream>>>(A1, b1, h1, N * 64, 64);

  // ---- convd1: hd1 = relu(agg(dat @ Wd1) + bd1), F=32, K=64 ----
  gemm_wmma_kernel<2><<<ggrid, gblk, 0, stream>>>(dat, Wpd1, Gd, nRowTiles, 64);
  selfinit_kernel<<<cdivll((long long)N * 32, T), T, 0, stream>>>(Gd, dinv2, Ad, N * 32, 32);
  scatter_kernel<32><<<cdivll((long long)E * 8, T), T, 0, stream>>>(Gd, src, dst, coef, Ad, E);
  bias_relu_kernel<<<cdivll((long long)N * 32, T), T, 0, stream>>>(Ad, bd1, hd1, N * 32, 32);

  // ---- conv2: A2 = agg(h1 @ W2)  (bias folded into final), F=16, K=64 ----
  gemm_wmma_kernel<1><<<ggrid, gblk, 0, stream>>>(h1, Wp2, G2, nRowTiles, 64);
  selfinit_kernel<<<cdivll((long long)N * 16, T), T, 0, stream>>>(G2, dinv2, A2, N * 16, 16);
  scatter_kernel<16><<<cdivll((long long)E * 4, T), T, 0, stream>>>(G2, src, dst, coef, A2, E);

  // ---- convd2: Ad2 = agg(hd1 @ Wd2), F=16, K=32 ----
  gemm_wmma_kernel<1><<<ggrid, gblk, 0, stream>>>(hd1, Wpd2, Gd2, nRowTiles, 32);
  selfinit_kernel<<<cdivll((long long)N * 16, T), T, 0, stream>>>(Gd2, dinv2, Ad2, N * 16, 16);
  scatter_kernel<16><<<cdivll((long long)E * 4, T), T, 0, stream>>>(Gd2, src, dst, coef, Ad2, E);

  // ---- combine + log_softmax ----
  final_kernel<<<cdivll(N, T), T, 0, stream>>>(A2, b2, Ad2, bd2, out, N);
}
